// vAttn_71073118814731
// MI455X (gfx1250) — matrix operand order repaired and checked by reading, not compiled, not run on hardware
//
#include <hip/hip_runtime.h>

#define DEV __device__ __forceinline__

typedef __attribute__((ext_vector_type(16))) __bf16 bf16x16;
typedef __attribute__((ext_vector_type(8)))  __bf16 bf16x8;
typedef __attribute__((ext_vector_type(8)))  float  f32x8;

// Problem dims (fixed by reference)
constexpr int Bz = 8, Cc = 256, Vv = 256, Tt = 64, Cq = 32, C1 = 128;
constexpr int SROW = Vv + 4;              // padded LDS row stride (floats) to dodge 64-bank conflicts

DEV unsigned short f2bf_us(float f) {
    unsigned u = __builtin_bit_cast(unsigned, f);
    unsigned r = u + 0x7FFFu + ((u >> 16) & 1u);     // round-to-nearest-even
    return (unsigned short)(r >> 16);
}
DEV __bf16 f2bf(float f) { return __builtin_bit_cast(__bf16, f2bf_us(f)); }

DEV f32x8 zero8() {
    f32x8 z;
    #pragma unroll
    for (int j = 0; j < 8; ++j) z[j] = 0.0f;
    return z;
}
DEV bf16x16 zero16b() {
    bf16x16 z;
    #pragma unroll
    for (int j = 0; j < 16; ++j) z[j] = f2bf(0.0f);
    return z;
}

// A-fragment (16x32 bf16, M x K): lane m = lane&15, group = lane>>4.
// halves h<8  -> K = 8*group + h          (contiguous 8)
// halves h>=8 -> K = 16 + 8*group + (h-8) (contiguous 8)
DEV bf16x16 load_a_frag(const unsigned short* rowK0, int group) {
    bf16x8 lo = *reinterpret_cast<const bf16x8*>(rowK0 + 8 * group);
    bf16x8 hi = *reinterpret_cast<const bf16x8*>(rowK0 + 16 + 8 * group);
    bf16x16 a;
    #pragma unroll
    for (int j = 0; j < 8; ++j) { a[j] = lo[j]; a[8 + j] = hi[j]; }
    return a;
}
// A-fragment sourced from f32 (LDS attention probabilities), converted to bf16
DEV bf16x16 load_a_frag_f32(const float* rowK0, int group) {
    bf16x16 a;
    #pragma unroll
    for (int j = 0; j < 8; ++j) {
        a[j]     = f2bf(rowK0[8 * group + j]);
        a[8 + j] = f2bf(rowK0[16 + 8 * group + j]);
    }
    return a;
}
// B-fragment (32x16 bf16, K x N): lane n = lane&15, group = lane>>4.
// halves h<8 -> K = 8*group + h; halves h>=8 -> K = 16 + 8*group + (h-8)
DEV bf16x16 load_b_frag(const unsigned short* colK0, int group) {
    bf16x8 lo = *reinterpret_cast<const bf16x8*>(colK0 + 8 * group);
    bf16x8 hi = *reinterpret_cast<const bf16x8*>(colK0 + 16 + 8 * group);
    bf16x16 b;
    #pragma unroll
    for (int j = 0; j < 8; ++j) { b[j] = lo[j]; b[8 + j] = hi[j]; }
    return b;
}
DEV f32x8 wmma_bf16(bf16x16 a, bf16x16 b, f32x8 c) {
    return __builtin_amdgcn_wmma_f32_16x16x32_bf16(false, a, false, b, (short)0, c, false, false);
}

// ---------------- Kernel 0: x [B,C,V,T] f32 -> xT [B,V,T,C] bf16 ----------------
__global__ __launch_bounds__(256) void k_cast_x(const float* __restrict__ x,
                                                unsigned short* __restrict__ xT) {
    size_t i = (size_t)blockIdx.x * blockDim.x + threadIdx.x;
    int t = (int)(i % Tt); size_t r = i / Tt;
    int v = (int)(r % Vv); r /= Vv;
    int c = (int)(r % Cc); int b = (int)(r / Cc);
    xT[(((size_t)b * Vv + v) * Tt + t) * Cc + c] = f2bf_us(x[i]);
}

// ---------------- Kernel 1: w [O,I,3,3] f32 -> wp [O][tap][I] bf16 ----------------
__global__ __launch_bounds__(256) void k_pack_w(const float* __restrict__ w,
                                                unsigned short* __restrict__ wp,
                                                int O, int I) {
    size_t i = (size_t)blockIdx.x * blockDim.x + threadIdx.x;
    if (i >= (size_t)O * I * 9) return;
    int kx = (int)(i % 3); size_t r = i / 3;
    int ky = (int)(r % 3); r /= 3;
    int ic = (int)(r % I); int o = (int)(r / I);
    wp[((size_t)o * 9 + (ky * 3 + kx)) * I + ic] = f2bf_us(w[i]);
}

// ---------------- Kernel 2: fused implicit-GEMM conv for q,k,v ----------------
// One wave per 16-pixel tile; 12 M-tiles (q:2, k:2, v:8) share every x B-fragment.
// A-fragments are batched 6-at-a-time so loads clause together and a single
// s_wait_loadcnt covers a run of independent WMMAs.
// q,k stored [B][T][V][Cq]; v stored transposed [B][T][C1][V] for the P·V GEMM.
__global__ __launch_bounds__(32) void k_conv_qkv(const unsigned short* __restrict__ xT,
                                                 const unsigned short* __restrict__ wqkv,
                                                 unsigned short* __restrict__ qb,
                                                 unsigned short* __restrict__ kb,
                                                 unsigned short* __restrict__ vb) {
    int ptile = blockIdx.x, b = blockIdx.y;
    int lane = threadIdx.x, group = lane >> 4, ln = lane & 15;
    int p0 = ptile * 16, v0 = p0 / Tt, t0 = p0 % Tt;  // 16 consecutive t at fixed v

    f32x8 acc[12];
    #pragma unroll
    for (int mt = 0; mt < 12; ++mt) acc[mt] = zero8();

    for (int tap = 0; tap < 9; ++tap) {
        int dy = tap / 3 - 1, dx = tap % 3 - 1;
        int vs = v0 + dy, ts = t0 + ln + dx;                 // ts is per-lane (pixel = N)
        bool inb = ((unsigned)vs < (unsigned)Vv) && ((unsigned)ts < (unsigned)Tt);
        const unsigned short* bbase = xT + (((size_t)b * Vv + vs) * Tt + ts) * Cc;
        for (int ct = 0; ct < 8; ++ct) {                     // K chunks over C=256
            bf16x16 bf = inb ? load_b_frag(bbase + ct * 32, group) : zero16b();
            #pragma unroll
            for (int half = 0; half < 2; ++half) {
                bf16x16 afr[6];
                #pragma unroll
                for (int j = 0; j < 6; ++j) {
                    int mt = half * 6 + j;
                    afr[j] = load_a_frag(
                        wqkv + (((size_t)(mt * 16 + ln) * 9) + tap) * Cc + ct * 32, group);
                }
                #pragma unroll
                for (int j = 0; j < 6; ++j) {
                    int mt = half * 6 + j;
                    acc[mt] = wmma_bf16(afr[j], bf, acc[mt]);
                }
            }
        }
    }
    // D layout: VGPR r -> (M = r + 8*group, N = ln)
    int t = t0 + ln;
    #pragma unroll
    for (int mt = 0; mt < 12; ++mt) {
        #pragma unroll
        for (int r = 0; r < 8; ++r) {
            int m = r + 8 * group;
            unsigned short hv = f2bf_us(acc[mt][r]);
            if (mt < 4) {
                unsigned short* ob = (mt < 2) ? qb : kb;
                int o = (mt & 1) * 16 + m;
                ob[(((size_t)b * Tt + t) * Vv + v0) * Cq + o] = hv;
            } else {
                int o = (mt - 4) * 16 + m;
                vb[(((size_t)b * Tt + t) * C1 + o) * Vv + v0] = hv;  // [b][t][c][u]
            }
        }
    }
}

// ---------------- Kernel 3: attention per (b,t) slice over V ----------------
// 8 waves; full 256x256 score matrix in padded LDS (SROW stride), wave32 softmax,
// then P·V with A-fragments converted f32->bf16 out of LDS.
__global__ __launch_bounds__(256) void k_attn(const unsigned short* __restrict__ qb,
                                              const unsigned short* __restrict__ kb,
                                              const unsigned short* __restrict__ vb,
                                              unsigned short* __restrict__ avT) {
    extern __shared__ float Ssm[];                     // SROW * 256 floats (~260 KB)
    int t = blockIdx.x, b = blockIdx.y;
    int tid = threadIdx.x, wave = tid >> 5, lane = tid & 31;
    int group = lane >> 4, ln = lane & 15;
    const size_t rowbase = ((size_t)b * Tt + t) * Vv;

    // ---- S = Q·K^T (Cq=32 => single WMMA per 16x16 tile) ----
    // K fragment is double-buffered so the next tile's loads overlap this WMMA.
    for (int vt = wave * 2; vt < wave * 2 + 2; ++vt) {
        int v0 = vt * 16;
        bf16x16 qf = load_a_frag(qb + (rowbase + v0 + ln) * Cq, group);
        bf16x16 kf = load_b_frag(kb + (rowbase + ln) * Cq, group);   // ut = 0
        for (int ut = 0; ut < 16; ++ut) {
            bf16x16 cur = kf;
            if (ut < 15)
                kf = load_b_frag(kb + (rowbase + (ut + 1) * 16 + ln) * Cq, group);
            f32x8 c = wmma_bf16(qf, cur, zero8());
            #pragma unroll
            for (int r = 0; r < 8; ++r)
                Ssm[(size_t)(v0 + r + 8 * group) * SROW + ut * 16 + ln] = c[r];
        }
    }
    __syncthreads();

    // ---- row softmax (32 rows per wave, 8 elems per lane, shuffle reductions) ----
    for (int row = wave; row < Vv; row += 8) {
        float* rp = Ssm + (size_t)row * SROW;
        float vals[8], mx = -3.0e38f;
        #pragma unroll
        for (int j = 0; j < 8; ++j) { vals[j] = rp[lane + 32 * j]; mx = fmaxf(mx, vals[j]); }
        for (int off = 16; off > 0; off >>= 1) mx = fmaxf(mx, __shfl_xor(mx, off, 32));
        float sum = 0.0f;
        #pragma unroll
        for (int j = 0; j < 8; ++j) { vals[j] = __expf(vals[j] - mx); sum += vals[j]; }
        for (int off = 16; off > 0; off >>= 1) sum += __shfl_xor(sum, off, 32);
        float inv = 1.0f / sum;
        #pragma unroll
        for (int j = 0; j < 8; ++j) rp[lane + 32 * j] = vals[j] * inv;
    }
    __syncthreads();

    // ---- AV = P·V ; av stored transposed [B][V][T][C1] for the wo conv ----
    // V fragments batched 4-at-a-time per K-chunk for clause-grouped loads.
    for (int vt = wave * 2; vt < wave * 2 + 2; ++vt) {
        int v0 = vt * 16;
        const float* prow = Ssm + (size_t)(v0 + ln) * SROW;
        f32x8 acc[8];
        #pragma unroll
        for (int nt = 0; nt < 8; ++nt) acc[nt] = zero8();
        for (int uc = 0; uc < 8; ++uc) {                       // K chunks over V=256
            bf16x16 pf = load_a_frag_f32(prow + uc * 32, group);
            #pragma unroll
            for (int g4 = 0; g4 < 2; ++g4) {
                bf16x16 vfr[4];
                #pragma unroll
                for (int j = 0; j < 4; ++j) {
                    int nt = g4 * 4 + j;
                    vfr[j] = load_b_frag(
                        vb + (((size_t)b * Tt + t) * C1 + nt * 16 + ln) * Vv + uc * 32, group);
                }
                #pragma unroll
                for (int j = 0; j < 4; ++j) {
                    int nt = g4 * 4 + j;
                    acc[nt] = wmma_bf16(pf, vfr[j], acc[nt]);
                }
            }
        }
        #pragma unroll
        for (int nt = 0; nt < 8; ++nt) {
            #pragma unroll
            for (int r = 0; r < 8; ++r) {
                int vvr = v0 + r + 8 * group;
                avT[(((size_t)b * Vv + vvr) * Tt + t) * C1 + nt * 16 + ln] =
                    f2bf_us(acc[nt][r]);
            }
        }
    }
}

// ---------------- Kernel 4: wo conv (implicit GEMM) + residual ----------------
// All 4 (A,B) fragment pairs of a tap are loaded before its 4 WMMAs.
__global__ __launch_bounds__(32) void k_conv_out(const unsigned short* __restrict__ avT,
                                                 const unsigned short* __restrict__ wop,
                                                 const float* __restrict__ x,
                                                 const float* __restrict__ sigma,
                                                 float* __restrict__ out) {
    int ptile = blockIdx.x, b = blockIdx.y, mt = blockIdx.z;
    int lane = threadIdx.x, group = lane >> 4, ln = lane & 15;
    int p0 = ptile * 16, v0 = p0 / Tt, t0 = p0 % Tt;

    f32x8 acc = zero8();
    for (int tap = 0; tap < 9; ++tap) {
        int dy = tap / 3 - 1, dx = tap % 3 - 1;
        int vs = v0 + dy, ts = t0 + ln + dx;
        bool inb = ((unsigned)vs < (unsigned)Vv) && ((unsigned)ts < (unsigned)Tt);
        const unsigned short* bbase = avT + (((size_t)b * Vv + vs) * Tt + ts) * C1;
        bf16x16 bfr[4], afr[4];
        #pragma unroll
        for (int ct = 0; ct < 4; ++ct) {                       // K chunks over C1=128
            bfr[ct] = inb ? load_b_frag(bbase + ct * 32, group) : zero16b();
            afr[ct] = load_a_frag(
                wop + (((size_t)(mt * 16 + ln) * 9) + tap) * C1 + ct * 32, group);
        }
        #pragma unroll
        for (int ct = 0; ct < 4; ++ct) acc = wmma_bf16(afr[ct], bfr[ct], acc);
    }
    float sg = sigma[0];
    int t = t0 + ln;
    #pragma unroll
    for (int r = 0; r < 8; ++r) {
        int o = mt * 16 + r + 8 * group;
        size_t idx = (((size_t)b * Cc + o) * Vv + v0) * Tt + t;
        out[idx] = x[idx] + sg * acc[r];
    }
}

extern "C" void kernel_launch(void* const* d_in, const int* in_sizes, int n_in,
                              void* d_out, int out_size, void* d_ws, size_t ws_size,
                              hipStream_t stream) {
    (void)in_sizes; (void)n_in; (void)out_size; (void)ws_size;
    const float* x  = (const float*)d_in[0];
    const float* wq = (const float*)d_in[1];
    const float* wk = (const float*)d_in[2];
    const float* wv = (const float*)d_in[3];
    const float* wo = (const float*)d_in[4];
    const float* sg = (const float*)d_in[5];
    float* out = (float*)d_out;

    // Workspace partition (bytes, 256-aligned sections)
    char* ws = (char*)d_ws;
    size_t off = 0;
    auto take = [&](size_t bytes) { char* p = ws + off; off = (off + bytes + 255) & ~(size_t)255; return p; };
    unsigned short* xT   = (unsigned short*)take((size_t)Bz * Vv * Tt * Cc * 2);   // 67 MB
    unsigned short* qb   = (unsigned short*)take((size_t)Bz * Tt * Vv * Cq * 2);   // 8.4 MB
    unsigned short* kb   = (unsigned short*)take((size_t)Bz * Tt * Vv * Cq * 2);   // 8.4 MB
    unsigned short* vb   = (unsigned short*)take((size_t)Bz * Tt * C1 * Vv * 2);   // 33.6 MB
    unsigned short* avT  = (unsigned short*)take((size_t)Bz * Vv * Tt * C1 * 2);   // 33.6 MB
    unsigned short* wqkv = (unsigned short*)take((size_t)(Cq + Cq + C1) * 9 * Cc * 2);
    unsigned short* wop  = (unsigned short*)take((size_t)Cc * 9 * C1 * 2);

    // 0) transpose/cast x
    {
        size_t n = (size_t)Bz * Cc * Vv * Tt;
        k_cast_x<<<dim3((unsigned)(n / 256)), 256, 0, stream>>>(x, xT);
    }
    // 1) pack weights: wq rows 0..31, wk rows 32..63, wv rows 64..191 of wqkv; wo -> wop
    k_pack_w<<<dim3((32 * 256 * 9 + 255) / 256), 256, 0, stream>>>(wq, wqkv, 32, 256);
    k_pack_w<<<dim3((32 * 256 * 9 + 255) / 256), 256, 0, stream>>>(wk, wqkv + (size_t)32 * 9 * 256, 32, 256);
    k_pack_w<<<dim3((128 * 256 * 9 + 255) / 256), 256, 0, stream>>>(wv, wqkv + (size_t)64 * 9 * 256, 128, 256);
    k_pack_w<<<dim3((256 * 128 * 9 + 255) / 256), 256, 0, stream>>>(wo, wop, 256, 128);

    // 2) fused q/k/v conv: 1024 pixel-tiles x 8 batches, one wave each
    k_conv_qkv<<<dim3(Vv * Tt / 16, Bz), 32, 0, stream>>>(xT, wqkv, qb, kb, vb);

    // 3) attention: one workgroup per (b,t); ~260 KB dynamic LDS for the score matrix
    k_attn<<<dim3(Tt, Bz), 256, (size_t)SROW * Vv * sizeof(float), stream>>>(qb, kb, vb, avT);

    // 4) wo conv + residual
    k_conv_out<<<dim3(Vv * Tt / 16, Bz, Cc / 16), 32, 0, stream>>>(avT, wop, x, sg, out);
}
